// GINConvNet_68169720922993
// MI455X (gfx1250) — matrix operand order repaired and checked by reading, not compiled, not run on hardware
//
#include <hip/hip_runtime.h>
#include <hip/hip_bf16.h>
#include <cstddef>

// ---------------------------------------------------------------------------
// Types for CDNA5 WMMA (wave32): v_wmma_f32_16x16x32_bf16
// ---------------------------------------------------------------------------
typedef __bf16 bf16_t;
typedef __attribute__((ext_vector_type(16))) __bf16 v16bf;
typedef __attribute__((ext_vector_type(8)))  __bf16 v8bf;
typedef __attribute__((ext_vector_type(8)))  float  v8f;

union B16 {            // 16 bf16 operand = 8 VGPRs
  v16bf v;
  v8bf  h[2];
  unsigned int u[8];
};

static __device__ inline bf16_t bfc(float f) { return (bf16_t)f; }

// ---------------------------------------------------------------------------
// Generic bf16 GEMM:  C[M,Nout] = A[M,Kpad] @ B  (+bias, +relu)
//   A   : bf16 row-major [M, Kpad]          (Kpad multiple of 32)
//   Bsw : bf16 pre-transposed [Nout, Kpad]  (Bsw[n][k] = Wmath[k][n])
//   Cf  : optional f32 out [M, Nout];  Cb: optional bf16 out, stride ldCb,
//         column offset colOff (used to write directly into concat buffers)
// One wave computes one 16x16 output tile; K=32 per wmma step.
// ---------------------------------------------------------------------------
__global__ void __launch_bounds__(256)
gemm_bf16(const bf16_t* __restrict__ A, const bf16_t* __restrict__ Bsw,
          const float* __restrict__ bias,
          float* __restrict__ Cf, bf16_t* __restrict__ Cb,
          int ldCb, int colOff, int M, int Kpad, int Nout, int relu)
{
  const int lane = threadIdx.x & 31;
  const int wid  = blockIdx.x * (blockDim.x >> 5) + (threadIdx.x >> 5);
  const int nTiles = Nout >> 4;
  const int mTiles = (M + 15) >> 4;
  const int mt = wid / nTiles;
  const int nt = wid - mt * nTiles;
  if (mt >= mTiles) return;                 // wave-uniform exit

  const int hf = lane >> 4;                 // K-half select (A/B operand layout)
  int rowA = mt * 16 + (lane & 15);
  if (rowA >= M) rowA = M - 1;              // clamp; stores are guarded
  const int gcol = nt * 16 + (lane & 15);

  const bf16_t* __restrict__ abase = A   + (size_t)rowA * Kpad + hf * 8;
  const bf16_t* __restrict__ bbase = Bsw + (size_t)gcol * Kpad + hf * 8;

  v8f acc = {};
  const int ksteps = Kpad >> 5;
  for (int k = 0; k < ksteps; ++k) {
    B16 a, b;
    // A: element e -> K = k*32 + hf*8 + e (e<8) ; k*32 + 16 + hf*8 + (e-8)
    a.h[0] = *(const v8bf*)(abase + k * 32);
    a.h[1] = *(const v8bf*)(abase + k * 32 + 16);
    // B: element e -> K = k*32 + hf*8 + e (e<8) ; k*32 + 16 + hf*8 + (e-8)
    b.h[0] = *(const v8bf*)(bbase + k * 32);
    b.h[1] = *(const v8bf*)(bbase + k * 32 + 16);
    acc = __builtin_amdgcn_wmma_f32_16x16x32_bf16(
        false, a.v, false, b.v, (short)0, acc, false, false);
  }

  const float bs = bias ? bias[gcol] : 0.0f;
  #pragma unroll
  for (int r = 0; r < 8; ++r) {
    const int row = mt * 16 + r + hf * 8;   // C/D: lanes>=16 hold M=r+8
    if (row >= M) continue;
    float v = acc[r] + bs;
    if (relu) v = fmaxf(v, 0.0f);
    if (Cf) Cf[(size_t)row * Nout + gcol] = v;
    if (Cb) Cb[(size_t)row * ldCb + colOff + gcol] = bfc(v);
  }
}

// ---------------------------------------------------------------------------
// Protein conv1d as implicit GEMM per graph:
//   out[b,o,w] = sum_{l<1500,kk<8} emb[t[b,l]][w+kk] * W[o,l,kk] + bias[o]
// M = w (121 -> 8 tiles), N = o (32 -> 2 tiles), K = 12000 (375 steps of 32).
// A operand gathered from LDS-staged emb table (dual copies for alignment),
// B operand streamed contiguously from bf16 conv weights (L2 resident).
// Output stored bf16 in reshape order [b][o*121 + w] for the fc1_xt GEMM.
// ---------------------------------------------------------------------------
__global__ void __launch_bounds__(512)
conv_protein(const int* __restrict__ target, const bf16_t* __restrict__ Wc,
             const unsigned short* __restrict__ embu,   // bf16 bits [26][128]
             const float* __restrict__ cbias, bf16_t* __restrict__ out)
{
  __shared__ unsigned short embA[26][136];  // natural copy (+zero pad)
  __shared__ unsigned short embB[26][136];  // shifted-by-one copy
  __shared__ int tseq[1500];

  const int b = blockIdx.x;
  for (int i = threadIdx.x; i < 26 * 136; i += 512) {
    const int v = i / 136, x = i - v * 136;
    embA[v][x] = (x < 128) ? embu[v * 128 + x] : (unsigned short)0;
    embB[v][x] = (x + 1 < 128) ? embu[v * 128 + x + 1] : (unsigned short)0;
  }
  for (int i = threadIdx.x; i < 1500; i += 512) tseq[i] = target[b * 1500 + i];
  __syncthreads();

  const int lane = threadIdx.x & 31;
  const int wid  = threadIdx.x >> 5;        // 16 waves = 8 m-tiles x 2 n-tiles
  const int mt = wid & 7, nt = wid >> 3;
  const int hf = lane >> 4;
  const int w  = mt * 16 + (lane & 15);     // output position (A row)
  const int o  = nt * 16 + (lane & 15);     // output channel (B col)

  const bf16_t* __restrict__ wbase = Wc + (size_t)o * 12000 + hf * 8;

  v8f acc = {};
  for (int ks = 0; ks < 375; ++ks) {
    if (ks + 8 < 375) __builtin_prefetch(wbase + (ks + 8) * 32, 0, 1);
    const int l0 = ks * 4 + hf;             // K = l*8+kk ; groups: l0 and l0+2
    const int v0 = tseq[l0];
    const int v1 = tseq[l0 + 2];
    B16 a;
    {
      const unsigned int* p = (const unsigned int*)
          ((w & 1) ? &embB[v0][w - 1] : &embA[v0][w]);   // 4B-aligned
      a.u[0] = p[0]; a.u[1] = p[1]; a.u[2] = p[2]; a.u[3] = p[3];
    }
    {
      const unsigned int* p = (const unsigned int*)
          ((w & 1) ? &embB[v1][w - 1] : &embA[v1][w]);
      a.u[4] = p[0]; a.u[5] = p[1]; a.u[6] = p[2]; a.u[7] = p[3];
    }
    B16 bm;
    bm.h[0] = *(const v8bf*)(wbase + ks * 32);
    bm.h[1] = *(const v8bf*)(wbase + ks * 32 + 16);
    acc = __builtin_amdgcn_wmma_f32_16x16x32_bf16(
        false, a.v, false, bm.v, (short)0, acc, false, false);
  }

  const float bs = cbias[o];
  #pragma unroll
  for (int r = 0; r < 8; ++r) {
    const int wo = mt * 16 + r + hf * 8;
    if (wo < 121) out[(size_t)b * 3872 + o * 121 + wo] = bfc(acc[r] + bs);
  }
}

// ---------------------------------------------------------------------------
// Edge aggregation: agg[dst] += u[src]  (32 floats/edge, 8 threads/edge)
// u and agg are L2-resident (12.8 MB each).
// ---------------------------------------------------------------------------
__global__ void edge_agg(const float* __restrict__ u, const int* __restrict__ src,
                         const int* __restrict__ dst, float* __restrict__ agg, int E)
{
  const long long tid = (long long)blockIdx.x * blockDim.x + threadIdx.x;
  const int e = (int)(tid >> 3);
  if (e >= E) return;
  const int c = ((int)tid & 7) * 4;
  const int s = src[e], d = dst[e];
  const float4 v = *(const float4*)(u + (size_t)s * 32 + c);
  float* o = agg + (size_t)d * 32 + c;
  atomicAdd(o + 0, v.x); atomicAdd(o + 1, v.y);
  atomicAdd(o + 2, v.z); atomicAdd(o + 3, v.w);
}

// t = relu(u + agg + ba)  -> bf16 (input to second GIN GEMM)
__global__ void post_agg(const float* __restrict__ u, const float* __restrict__ agg,
                         const float* __restrict__ ba, bf16_t* __restrict__ tb, int total)
{
  const int i = blockIdx.x * blockDim.x + threadIdx.x;
  if (i >= total) return;
  const int c = i & 31;
  tb[i] = bfc(fmaxf(u[i] + agg[i] + ba[c], 0.0f));
}

// h = bn(relu(v + bb)); optionally pool into hg via batch[] (layer 5)
__global__ void gin_finish(const float* __restrict__ v2, const float* __restrict__ bb,
                           const float* __restrict__ gam, const float* __restrict__ bet,
                           const float* __restrict__ mu, const float* __restrict__ var,
                           bf16_t* __restrict__ hb, const int* __restrict__ batch,
                           float* __restrict__ hg, int pool, int total)
{
  const int i = blockIdx.x * blockDim.x + threadIdx.x;
  if (i >= total) return;
  const int c = i & 31, n = i >> 5;
  float h = fmaxf(v2[i] + bb[c], 0.0f);
  h = (h - mu[c]) * (gam[c] * rsqrtf(var[c] + 1e-5f)) + bet[c];
  hb[i] = bfc(h);
  if (pool) atomicAdd(&hg[(size_t)batch[n] * 32 + c], h);
}

// ---------------------------------------------------------------------------
// Conversion / layout kernels
// ---------------------------------------------------------------------------
__global__ void f2bf(const float* __restrict__ s, bf16_t* __restrict__ d, int n) {
  const int i = blockIdx.x * blockDim.x + threadIdx.x;
  if (i < n) d[i] = bfc(s[i]);
}
// rows [N,K] f32 -> [N,Kpad] bf16, zero-padded K..Kpad
__global__ void pad_rows_bf(const float* __restrict__ s, bf16_t* __restrict__ d,
                            int N, int K, int Kpad) {
  const long long i = (long long)blockIdx.x * blockDim.x + threadIdx.x;
  if (i >= (long long)N * Kpad) return;
  const int n = (int)(i / Kpad), k = (int)(i - (long long)n * Kpad);
  d[i] = (k < K) ? bfc(s[(size_t)n * K + k]) : bfc(0.0f);
}
// W [K,Nout] f32 -> Wsw [Nout,Kpad] bf16 (transpose + pad)
__global__ void transpose_bf(const float* __restrict__ s, bf16_t* __restrict__ d,
                             int K, int Kpad, int Nout) {
  const long long i = (long long)blockIdx.x * blockDim.x + threadIdx.x;
  if (i >= (long long)Nout * Kpad) return;
  const int n = (int)(i / Kpad), k = (int)(i - (long long)n * Kpad);
  d[i] = (k < K) ? bfc(s[(size_t)k * Nout + n]) : bfc(0.0f);
}
__global__ void zerof(float* __restrict__ p, int n) {
  const int i = blockIdx.x * blockDim.x + threadIdx.x;
  if (i < n) p[i] = 0.0f;
}

// out[b] = dot(y2_bf[b,:256], out_w) + out_b
__global__ void final_head(const bf16_t* __restrict__ y2, const float* __restrict__ ow,
                           const float* __restrict__ ob, float* __restrict__ out, int B)
{
  const int b = blockIdx.x * blockDim.x + threadIdx.x;
  if (b >= B) return;
  float s = ob[0];
  const bf16_t* r = y2 + (size_t)b * 256;
  #pragma unroll 8
  for (int c = 0; c < 256; ++c) s += (float)r[c] * ow[c];
  out[b] = s;
}

// ---------------------------------------------------------------------------
extern "C" void kernel_launch(void* const* d_in, const int* in_sizes, int n_in,
                              void* d_out, int out_size, void* d_ws, size_t ws_size,
                              hipStream_t stream)
{
  (void)n_in; (void)ws_size;
  // ---- inputs (setup_inputs order) ----
  const float* x        = (const float*)d_in[0];
  const int*   eidx     = (const int*)d_in[1];
  const int*   batch    = (const int*)d_in[2];
  const int*   target   = (const int*)d_in[3];
  const float* g_w1a    = (const float*)d_in[4];
  const float* g_b1a    = (const float*)d_in[5];
  const float* g_w1b    = (const float*)d_in[6];
  const float* g_b1b    = (const float*)d_in[7];
  const float* gW_a     = (const float*)d_in[8];
  const float* gb_a     = (const float*)d_in[9];
  const float* gW_b     = (const float*)d_in[10];
  const float* gb_b     = (const float*)d_in[11];
  const float* bn_g     = (const float*)d_in[12];
  const float* bn_b     = (const float*)d_in[13];
  const float* bn_m     = (const float*)d_in[14];
  const float* bn_v     = (const float*)d_in[15];
  const float* fc1xd_w  = (const float*)d_in[16];
  const float* fc1xd_b  = (const float*)d_in[17];
  const float* emb      = (const float*)d_in[18];
  const float* convw    = (const float*)d_in[19];
  const float* convb    = (const float*)d_in[20];
  const float* fc1xt_w  = (const float*)d_in[21];
  const float* fc1xt_b  = (const float*)d_in[22];
  const float* fc1_w    = (const float*)d_in[23];
  const float* fc1_b    = (const float*)d_in[24];
  const float* fc2_w    = (const float*)d_in[25];
  const float* fc2_b    = (const float*)d_in[26];
  const float* out_w    = (const float*)d_in[27];
  const float* out_b    = (const float*)d_in[28];

  const int N = in_sizes[0] / 78;
  const int E = in_sizes[1] / 2;
  const int B = in_sizes[3] / 1500;
  const int* src = eidx;
  const int* dst = eidx + E;
  (void)out_size;

  // ---- workspace carve-up ----
  size_t off = 0;
  auto alloc = [&](size_t bytes) -> char* {
    char* p = (char*)d_ws + off;
    off += (bytes + 255) & ~(size_t)255;
    return p;
  };
  bf16_t* x_bf     = (bf16_t*)alloc((size_t)N * 96 * 2);
  bf16_t* h_bf     = (bf16_t*)alloc((size_t)N * 32 * 2);
  bf16_t* t_bf     = (bf16_t*)alloc((size_t)N * 32 * 2);
  float*  u        = (float*) alloc((size_t)N * 32 * 4);
  float*  agg      = (float*) alloc((size_t)N * 32 * 4);
  bf16_t* w1a_sw   = (bf16_t*)alloc(32 * 96 * 2);
  bf16_t* w1b_sw   = (bf16_t*)alloc(32 * 32 * 2);
  bf16_t* gwa_sw   = (bf16_t*)alloc(4 * 32 * 32 * 2);
  bf16_t* gwb_sw   = (bf16_t*)alloc(4 * 32 * 32 * 2);
  bf16_t* fcxd_sw  = (bf16_t*)alloc(128 * 32 * 2);
  bf16_t* convw_bf = (bf16_t*)alloc(32 * 12000 * 2);
  bf16_t* emb_bf   = (bf16_t*)alloc(26 * 128 * 2);
  bf16_t* convo_bf = (bf16_t*)alloc((size_t)B * 3872 * 2);
  bf16_t* fcxt_sw  = (bf16_t*)alloc((size_t)128 * 3872 * 2);
  bf16_t* fc1_sw   = (bf16_t*)alloc((size_t)1024 * 256 * 2);
  bf16_t* fc2_sw   = (bf16_t*)alloc((size_t)256 * 1024 * 2);
  float*  hg       = (float*) alloc((size_t)B * 32 * 4);
  bf16_t* hg_bf    = (bf16_t*)alloc((size_t)B * 32 * 2);
  bf16_t* xc_bf    = (bf16_t*)alloc((size_t)B * 256 * 2);
  bf16_t* y1_bf    = (bf16_t*)alloc((size_t)B * 1024 * 2);
  bf16_t* y2_bf    = (bf16_t*)alloc((size_t)B * 256 * 2);

  auto blk = [](long long total, int tpb) { return (unsigned)((total + tpb - 1) / tpb); };
  auto gemm = [&](const bf16_t* A, const bf16_t* Bsw, const float* bias, float* Cf,
                  bf16_t* Cb, int ldCb, int colOff, int M, int Kpad, int Nout, int relu) {
    const int waves = ((M + 15) / 16) * (Nout / 16);
    gemm_bf16<<<blk(waves, 8), 256, 0, stream>>>(A, Bsw, bias, Cf, Cb, ldCb, colOff,
                                                 M, Kpad, Nout, relu);
  };

  // ---- weight/input conversions (deterministic pre-pass) ----
  pad_rows_bf<<<blk((long long)N * 96, 256), 256, 0, stream>>>(x, x_bf, N, 78, 96);
  transpose_bf<<<blk(32 * 96, 256), 256, 0, stream>>>(g_w1a, w1a_sw, 78, 96, 32);
  transpose_bf<<<blk(32 * 32, 256), 256, 0, stream>>>(g_w1b, w1b_sw, 32, 32, 32);
  for (int i = 0; i < 4; ++i) {
    transpose_bf<<<blk(32 * 32, 256), 256, 0, stream>>>(gW_a + i * 1024, gwa_sw + i * 1024, 32, 32, 32);
    transpose_bf<<<blk(32 * 32, 256), 256, 0, stream>>>(gW_b + i * 1024, gwb_sw + i * 1024, 32, 32, 32);
  }
  transpose_bf<<<blk(128 * 32, 256), 256, 0, stream>>>(fc1xd_w, fcxd_sw, 32, 32, 128);
  f2bf<<<blk(32 * 12000, 256), 256, 0, stream>>>(convw, convw_bf, 32 * 12000);
  f2bf<<<blk(26 * 128, 256), 256, 0, stream>>>(emb, emb_bf, 26 * 128);
  transpose_bf<<<blk((long long)128 * 3872, 256), 256, 0, stream>>>(fc1xt_w, fcxt_sw, 3872, 3872, 128);
  transpose_bf<<<blk((long long)1024 * 256, 256), 256, 0, stream>>>(fc1_w, fc1_sw, 256, 256, 1024);
  transpose_bf<<<blk((long long)256 * 1024, 256), 256, 0, stream>>>(fc2_w, fc2_sw, 1024, 1024, 256);
  zerof<<<blk(B * 32, 256), 256, 0, stream>>>(hg, B * 32);

  // ---- GIN stack (segment_sum hoisted past the first linear layer) ----
  const int nc = N * 32;
  // layer 1
  gemm(x_bf, w1a_sw, nullptr, u, nullptr, 0, 0, N, 96, 32, 0);
  zerof<<<blk(nc, 256), 256, 0, stream>>>(agg, nc);
  edge_agg<<<blk((long long)E * 8, 256), 256, 0, stream>>>(u, src, dst, agg, E);
  post_agg<<<blk(nc, 256), 256, 0, stream>>>(u, agg, g_b1a, t_bf, nc);
  gemm(t_bf, w1b_sw, nullptr, u, nullptr, 0, 0, N, 32, 32, 0);
  gin_finish<<<blk(nc, 256), 256, 0, stream>>>(u, g_b1b, bn_g, bn_b, bn_m, bn_v,
                                               h_bf, nullptr, nullptr, 0, nc);
  // layers 2..5
  for (int i = 0; i < 4; ++i) {
    const int pool = (i == 3);
    gemm(h_bf, gwa_sw + i * 1024, nullptr, u, nullptr, 0, 0, N, 32, 32, 0);
    zerof<<<blk(nc, 256), 256, 0, stream>>>(agg, nc);
    edge_agg<<<blk((long long)E * 8, 256), 256, 0, stream>>>(u, src, dst, agg, E);
    post_agg<<<blk(nc, 256), 256, 0, stream>>>(u, agg, gb_a + i * 32, t_bf, nc);
    gemm(t_bf, gwb_sw + i * 1024, nullptr, u, nullptr, 0, 0, N, 32, 32, 0);
    gin_finish<<<blk(nc, 256), 256, 0, stream>>>(u, gb_b + i * 32,
        bn_g + (i + 1) * 32, bn_b + (i + 1) * 32, bn_m + (i + 1) * 32, bn_v + (i + 1) * 32,
        h_bf, batch, hg, pool, nc);
  }

  // ---- graph head: xd = relu(hg @ fc1_xd_w + b) -> xc[:, 0:128] ----
  f2bf<<<blk(B * 32, 256), 256, 0, stream>>>(hg, hg_bf, B * 32);
  gemm(hg_bf, fcxd_sw, fc1xd_b, nullptr, xc_bf, 256, 0, B, 32, 128, 1);

  // ---- protein branch: conv (implicit GEMM) then fc1_xt -> xc[:, 128:256] ----
  conv_protein<<<B, 512, 0, stream>>>(target, convw_bf,
                                      (const unsigned short*)emb_bf, convb, convo_bf);
  gemm(convo_bf, fcxt_sw, fc1xt_b, nullptr, xc_bf, 256, 128, B, 3872, 128, 0);

  // ---- MLP head ----
  gemm(xc_bf, fc1_sw, fc1_b, nullptr, y1_bf, 1024, 0, B, 256, 1024, 1);
  gemm(y1_bf, fc2_sw, fc2_b, nullptr, y2_bf, 256, 0, B, 1024, 256, 1);
  final_head<<<blk(B, 256), 256, 0, stream>>>(y2_bf, out_w, out_b, (float*)d_out, B);
}
